// Speller_57019985822507
// MI455X (gfx1250) — matrix operand order repaired and checked by reading, not compiled, not run on hardware
//
#include <hip/hip_runtime.h>

typedef _Float16 h16;
typedef __attribute__((ext_vector_type(16))) _Float16 v16h;
typedef __attribute__((ext_vector_type(8)))  _Float16 v8h;
typedef __attribute__((ext_vector_type(8)))  float    v8f;

#define B_   512
#define T_   256
#define V_   1000
#define VP_  1024
#define E_   256
#define H_   512
#define G4_  2048   /* 4*H */
#define LDP_ 1032   /* padded LDS row stride (floats) to break bank conflicts */

// ---------------- WMMA fragment helpers (wave32, 16x16x32 f16) ----------------
// Pointer-based loaders: caller precomputes the per-lane pointer once and
// advances it by 32 halfs per K-step; tile/gate selection via constant offsets.

// A fragment from p = base + (row0 + lane%16)*ld + k0 + (lane/16)*16.
// elems 0..15 = p[0..15]  (32B contiguous)
__device__ inline v16h ld_a(const h16* __restrict__ p) {
  return *(const v16h*)p;
}

// B fragment (B = W^T) from p = W + (n0 + lane%16)*ld + k0 + (lane/16)*16.
__device__ inline v16h ld_b(const h16* __restrict__ p) {
  return *(const v16h*)p;  // 32B contiguous
}

__device__ inline v8f wmma_f16(v16h a, v16h b, v8f c) {
  return __builtin_amdgcn_wmma_f32_16x16x32_f16(false, a, false, b, (short)0, c,
                                                false, false);
}

__device__ inline float sigmoidf_(float x) { return 1.f / (1.f + __expf(-x)); }

// Hardware tanh (CDNA5 TRANS op) with graceful fallbacks.
__device__ inline float fast_tanh(float x) {
#if __has_builtin(__builtin_amdgcn_tanhf)
  return __builtin_amdgcn_tanhf(x);
#elif __has_builtin(__builtin_amdgcn_tanh_f32)
  return __builtin_amdgcn_tanh_f32(x);
#else
  float y;
  asm volatile("v_tanh_f32 %0, %1" : "=v"(y) : "v"(x));
  return y;
#endif
}

// ---------------- prep kernels ----------------

// Xe[t][b][e] = emb[chars[b][t]][e] as f16, chars[b][0]=EOS(0), chars[b][t]=gt[b][t-1]
__global__ void embed_kernel(const int* __restrict__ gt, const float* __restrict__ emb,
                             h16* __restrict__ Xe) {
  size_t idx = (size_t)blockIdx.x * blockDim.x + threadIdx.x;  // < T*B*E
  int e = (int)(idx % E_);
  size_t tb = idx / E_;
  int b = (int)(tb % B_);
  int t = (int)(tb / B_);
  int ch = (t == 0) ? 0 : gt[(size_t)b * T_ + (t - 1)];
  Xe[idx] = (h16)emb[(size_t)ch * E_ + e];
}

__global__ void cvt_kernel(const float* __restrict__ src, h16* __restrict__ dst, int n) {
  int i = blockIdx.x * blockDim.x + threadIdx.x;
  if (i < n) dst[i] = (h16)src[i];
}

__global__ void bsum_kernel(const float* __restrict__ a, const float* __restrict__ b,
                            float* __restrict__ o, int n) {
  int i = blockIdx.x * blockDim.x + threadIdx.x;
  if (i < n) o[i] = a[i] + b[i];
}

// ---------------- fused LSTM-cell step ----------------
// gates = [x_t | h_cell] @ [W_ih | W_hh]^T + bias, fused with elementwise LSTM.
// Wave tile: 32 rows x 16 j-cols, N-tiles at {0,H,2H,3H}+j0 so each lane holds
// matching (i,f,g,o) for its (row, j). All 8 waves of a block share j0 (same B
// fragment addresses -> WGP$ hits); they cover 8 consecutive 32-row M groups.
// h double-buffered; c in place (per-lane exclusive). grid: (64, 2) x 256.
__global__ void __launch_bounds__(256) cell_step_kernel(
    int t, const h16* __restrict__ Xe,
    const h16* __restrict__ Hcur, h16* __restrict__ Hnxt,
    float* __restrict__ Cbuf,
    const h16* __restrict__ Wih0, const h16* __restrict__ Whh0,
    const h16* __restrict__ Wih1, const h16* __restrict__ Whh1,
    const float* __restrict__ bsum0, const float* __restrict__ bsum1) {
  int cell = blockIdx.y;
  const h16* Wih = cell ? Wih1 : Wih0;
  const h16* Whh = cell ? Whh1 : Whh0;
  const float* bs = cell ? bsum1 : bsum0;
  const h16* X  = Xe + (size_t)t * B_ * E_;
  const h16* Hp = Hcur + (size_t)cell * B_ * H_;
  h16*       Hn = Hnxt + (size_t)cell * B_ * H_;
  float*     Cm = Cbuf + (size_t)cell * B_ * H_;

  int widx = threadIdx.x >> 5;                  // wave within block: 0..7
  int jt   = blockIdx.x & 31;                   // j tile shared by whole block
  int mt   = (blockIdx.x >> 5) * 8 + widx;      // 0..15 -> 32-row group
  int row0 = mt * 32;
  int j0   = jt * 16;

  int lane = threadIdx.x & 31;
  int m = lane & 15, hi = lane >> 4;            // m doubles as n for B frags

  v8f acc[2][4] = {};

  {  // embedding part of K (ld = E_)
    const h16* pa = X + (size_t)(row0 + m) * E_ + hi * 16;
    const h16* pb = Wih + (size_t)(j0 + m) * E_ + hi * 16;
#pragma unroll 1
    for (int it = 0; it < E_ / 32; ++it) {
      v16h a0 = ld_a(pa);
      v16h a1 = ld_a(pa + 16 * E_);
#pragma unroll
      for (int g = 0; g < 4; ++g) {
        v16h b = ld_b(pb + (size_t)g * (H_ * E_));
        acc[0][g] = wmma_f16(a0, b, acc[0][g]);
        acc[1][g] = wmma_f16(a1, b, acc[1][g]);
      }
      pa += 32;
      pb += 32;
    }
  }
  {  // hidden part of K (ld = H_)
    const h16* pa = Hp + (size_t)(row0 + m) * H_ + hi * 16;
    const h16* pb = Whh + (size_t)(j0 + m) * H_ + hi * 16;
#pragma unroll 1
    for (int it = 0; it < H_ / 32; ++it) {
      v16h a0 = ld_a(pa);
      v16h a1 = ld_a(pa + 16 * H_);
#pragma unroll
      for (int g = 0; g < 4; ++g) {
        v16h b = ld_b(pb + (size_t)g * (H_ * H_));
        acc[0][g] = wmma_f16(a0, b, acc[0][g]);
        acc[1][g] = wmma_f16(a1, b, acc[1][g]);
      }
      pa += 32;
      pb += 32;
    }
  }

  // fused LSTM epilogue (lane owns cols j0+m, rows row0 + mi*16 + r + 8*hi)
  float bi = bs[j0 + m];
  float bf = bs[H_ + j0 + m];
  float bg = bs[2 * H_ + j0 + m];
  float bo = bs[3 * H_ + j0 + m];
  float* cp = Cm + (size_t)(row0 + 8 * hi) * H_ + j0 + m;
  h16*   hp = Hn + (size_t)(row0 + 8 * hi) * H_ + j0 + m;
#pragma unroll
  for (int mi = 0; mi < 2; ++mi) {
#pragma unroll
    for (int r = 0; r < 8; ++r) {
      size_t off = (size_t)(mi * 16 + r) * H_;
      float i = sigmoidf_(acc[mi][0][r] + bi);
      float f = sigmoidf_(acc[mi][1][r] + bf);
      float g = fast_tanh(acc[mi][2][r] + bg);
      float o = sigmoidf_(acc[mi][3][r] + bo);
      float c = f * cp[off] + i * g;
      cp[off] = c;
      hp[off] = (h16)(o * fast_tanh(c));
    }
  }
}

// logits = h1 @ W_out^T + b_out, then row-wise log_softmax, write out[b][t][:].
// One block = 16 rows x 1024 (padded V) cols, logits staged in LDS.
__global__ void __launch_bounds__(256) logits_kernel(
    int t, const h16* __restrict__ H1, const h16* __restrict__ Wout,
    const float* __restrict__ bout, float* __restrict__ out) {
  __shared__ float lt[16 * LDP_];
  int row0 = blockIdx.x * 16;
  int wid = threadIdx.x >> 5;
  int lane = threadIdx.x & 31;
  int n0w = wid * 128;  // each of 8 waves owns 8 N-tiles
  int m = lane & 15, hi = lane >> 4;

  v8f acc[8] = {};
  {
    const h16* pa = H1 + (size_t)(row0 + m) * H_ + hi * 16;
    const h16* pb = Wout + (size_t)(n0w + m) * H_ + hi * 16;
#pragma unroll 1
    for (int it = 0; it < H_ / 32; ++it) {
      v16h a = ld_a(pa);
#pragma unroll
      for (int j = 0; j < 8; ++j) {
        v16h b = ld_b(pb + (size_t)j * (16 * H_));
        acc[j] = wmma_f16(a, b, acc[j]);
      }
      pa += 32;
      pb += 32;
    }
  }
#pragma unroll
  for (int j = 0; j < 8; ++j) {
    int col = n0w + j * 16 + m;
    bool pad = (col >= V_);
    float bb = pad ? 0.f : bout[col];
#pragma unroll
    for (int r = 0; r < 8; ++r)
      lt[(r + 8 * hi) * LDP_ + col] = pad ? -1e30f : (acc[j][r] + bb);
  }
  __syncthreads();

  // wave wid handles rows 2*wid and 2*wid+1 (wave32 shuffle reductions)
#pragma unroll 1
  for (int rr = 0; rr < 2; ++rr) {
    int r = wid * 2 + rr;
    const float* Lr = &lt[r * LDP_];
    float mx = -1e30f;
    for (int v = lane; v < VP_; v += 32) mx = fmaxf(mx, Lr[v]);
#pragma unroll
    for (int s = 16; s > 0; s >>= 1) mx = fmaxf(mx, __shfl_xor(mx, s, 32));
    float se = 0.f;
    for (int v = lane; v < VP_; v += 32) se += __expf(Lr[v] - mx);
#pragma unroll
    for (int s = 16; s > 0; s >>= 1) se += __shfl_xor(se, s, 32);
    float lse = mx + __logf(se);
    float* Or = out + ((size_t)(row0 + r) * T_ + t) * V_;
    for (int v = lane; v < V_; v += 32) Or[v] = Lr[v] - lse;
  }
}

// ---------------- driver ----------------

extern "C" void kernel_launch(void* const* d_in, const int* in_sizes, int n_in,
                              void* d_out, int out_size, void* d_ws, size_t ws_size,
                              hipStream_t stream) {
  (void)in_sizes; (void)n_in; (void)out_size; (void)ws_size;
  const int*   gt    = (const int*)d_in[1];
  const float* emb   = (const float*)d_in[2];
  const float* Wih0f = (const float*)d_in[3];
  const float* Whh0f = (const float*)d_in[4];
  const float* bih0  = (const float*)d_in[5];
  const float* bhh0  = (const float*)d_in[6];
  const float* Wih1f = (const float*)d_in[7];
  const float* Whh1f = (const float*)d_in[8];
  const float* bih1  = (const float*)d_in[9];
  const float* bhh1  = (const float*)d_in[10];
  const float* Woutf = (const float*)d_in[11];
  const float* bout  = (const float*)d_in[12];
  float* out = (float*)d_out;

  char* ws = (char*)d_ws;
  size_t off = 0;
  auto alloc = [&](size_t bytes) -> void* {
    void* p = ws + off;
    off = (off + bytes + 255) & ~(size_t)255;
    return p;
  };
  h16*   Xe    = (h16*)  alloc((size_t)T_ * B_ * E_ * 2);  // 67 MB
  h16*   Wih0h = (h16*)  alloc((size_t)G4_ * E_ * 2);
  h16*   Whh0h = (h16*)  alloc((size_t)G4_ * H_ * 2);
  h16*   Wih1h = (h16*)  alloc((size_t)G4_ * E_ * 2);
  h16*   Whh1h = (h16*)  alloc((size_t)G4_ * H_ * 2);
  h16*   Wouth = (h16*)  alloc((size_t)VP_ * H_ * 2);
  float* bsum0 = (float*)alloc((size_t)G4_ * 4);
  float* bsum1 = (float*)alloc((size_t)G4_ * 4);
  h16*   Hping = (h16*)  alloc((size_t)2 * B_ * H_ * 2);
  h16*   Hpong = (h16*)  alloc((size_t)2 * B_ * H_ * 2);
  float* Cbuf  = (float*)alloc((size_t)2 * B_ * H_ * 4);

  // deterministic re-init each call (graph-capturable async ops only)
  hipMemsetAsync(Hping, 0, (size_t)2 * B_ * H_ * 2, stream);
  hipMemsetAsync(Cbuf, 0, (size_t)2 * B_ * H_ * 4, stream);
  hipMemsetAsync(Wouth, 0, (size_t)VP_ * H_ * 2, stream);  // zero-pad rows 1000..1023

  embed_kernel<<<(T_ * B_ * E_) / 256, 256, 0, stream>>>(gt, emb, Xe);
  cvt_kernel<<<(G4_ * E_ + 255) / 256, 256, 0, stream>>>(Wih0f, Wih0h, G4_ * E_);
  cvt_kernel<<<(G4_ * H_ + 255) / 256, 256, 0, stream>>>(Whh0f, Whh0h, G4_ * H_);
  cvt_kernel<<<(G4_ * E_ + 255) / 256, 256, 0, stream>>>(Wih1f, Wih1h, G4_ * E_);
  cvt_kernel<<<(G4_ * H_ + 255) / 256, 256, 0, stream>>>(Whh1f, Whh1h, G4_ * H_);
  cvt_kernel<<<(V_ * H_ + 255) / 256, 256, 0, stream>>>(Woutf, Wouth, V_ * H_);
  bsum_kernel<<<(G4_ + 255) / 256, 256, 0, stream>>>(bih0, bhh0, bsum0, G4_);
  bsum_kernel<<<(G4_ + 255) / 256, 256, 0, stream>>>(bih1, bhh1, bsum1, G4_);

  h16* Hcur = Hping;
  h16* Hnxt = Hpong;
  for (int t = 0; t < T_; ++t) {
    cell_step_kernel<<<dim3(64, 2, 1), 256, 0, stream>>>(
        t, Xe, Hcur, Hnxt, Cbuf, Wih0h, Whh0h, Wih1h, Whh1h, bsum0, bsum1);
    logits_kernel<<<dim3(B_ / 16, 1, 1), 256, 0, stream>>>(
        t, Hnxt + (size_t)B_ * H_, Wouth, bout, out);
    h16* tmp = Hcur; Hcur = Hnxt; Hnxt = tmp;
  }
}
